// MHA_52312701665688
// MI455X (gfx1250) — compile-verified, matrix operand order checked by reading, not run on hardware
//
#include <hip/hip_runtime.h>

// ---------------------------------------------------------------------------
// MHA forward for MI455X (gfx1250): bf16 WMMA everywhere, flash attention,
// software-pipelined fragment loads so WMMAs are not gated on loadcnt==0.
// T=2048, C=1024, H=16, D=64.
// ---------------------------------------------------------------------------

#define DEV __device__ __forceinline__

typedef __bf16 v16bf __attribute__((ext_vector_type(16)));
typedef float  v8f   __attribute__((ext_vector_type(8)));
typedef unsigned int u32x4 __attribute__((ext_vector_type(4)));
typedef unsigned int u32x8 __attribute__((ext_vector_type(8)));

static constexpr int T  = 2048;
static constexpr int C  = 1024;
static constexpr int H  = 16;
static constexpr int DH = 64;      // head dim
static constexpr int C3 = 3 * C;   // 3072

// float -> bf16 (round to nearest even)
DEV unsigned short f2bf(float f) {
    unsigned int u = __builtin_bit_cast(unsigned int, f);
    unsigned int r = u + 0x7FFFu + ((u >> 16) & 1u);
    return (unsigned short)(r >> 16);
}

// Load one 16x32 bf16 A/B fragment row/column for this lane.
// CDNA5 16-bit A layout: lane holds two contiguous 8-element K-runs at
// offsets 8*(lane>>4) and 16+8*(lane>>4). Two b128 loads.
DEV v16bf load_frag(const unsigned short* base, int lhi) {
    u32x4 a = *reinterpret_cast<const u32x4*>(base + 8 * lhi);
    u32x4 b = *reinterpret_cast<const u32x4*>(base + 16 + 8 * lhi);
    u32x8 w;
    w[0] = a[0]; w[1] = a[1]; w[2] = a[2]; w[3] = a[3];
    w[4] = b[0]; w[5] = b[1]; w[6] = b[2]; w[7] = b[3];
    return __builtin_bit_cast(v16bf, w);
}

DEV v8f wmma_bf16(v16bf a, v16bf b, v8f c) {
    // D = A(16x32 bf16) * B(32x16 bf16) + C(16x16 f32)
    return __builtin_amdgcn_wmma_f32_16x16x32_bf16(
        /*neg_a=*/false, a, /*neg_b=*/false, b,
        /*c_mod=*/(short)0, c, /*reuse_a=*/false, /*reuse_b=*/false);
}

// ---------------------------------------------------------------------------
// Conversion kernels
// ---------------------------------------------------------------------------
__global__ void cvt_f32_bf16(const float* __restrict__ src,
                             unsigned short* __restrict__ dst, int n) {
    int i = blockIdx.x * blockDim.x + threadIdx.x;
    if (i < n) dst[i] = f2bf(src[i]);
}

// dst[c][r] = bf16(src[r][c])  (src is rows x cols, row-major)
__global__ void cvt_transpose_f32_bf16(const float* __restrict__ src,
                                       unsigned short* __restrict__ dst,
                                       int rows, int cols) {
    int i = blockIdx.x * blockDim.x + threadIdx.x;
    if (i >= rows * cols) return;
    int r = i / cols;
    int c = i - r * cols;
    dst[c * rows + r] = f2bf(src[i]);
}

// ---------------------------------------------------------------------------
// GEMM core: one wave computes a 32x64 strip of A(32xK) * Bt^T, K = C.
// A:  [M][K] bf16 row-major.  Bt: [N][K] bf16 (B transposed) so both
// fragment types are contiguous b128 loads. Double-buffered: next K-step's
// fragments are in flight while the current 8 WMMAs execute.
// ---------------------------------------------------------------------------
DEV void gemm_core_2x4(const unsigned short* __restrict__ A,
                       const unsigned short* __restrict__ Bt,
                       int tm32, int tn64, int lm, int lhi, v8f acc[2][4]) {
    const unsigned short* Ar0 = A + (size_t)(tm32 * 32 + lm) * C;
    const unsigned short* Ar1 = A + (size_t)(tm32 * 32 + 16 + lm) * C;
    const unsigned short* Br[4];
#pragma unroll
    for (int s = 0; s < 4; ++s)
        Br[s] = Bt + (size_t)(tn64 * 64 + s * 16 + lm) * C;

    // prologue: fragments for k0 = 0
    v16bf a0 = load_frag(Ar0, lhi);
    v16bf a1 = load_frag(Ar1, lhi);
    v16bf b0 = load_frag(Br[0], lhi);
    v16bf b1 = load_frag(Br[1], lhi);
    v16bf b2 = load_frag(Br[2], lhi);
    v16bf b3 = load_frag(Br[3], lhi);

#pragma unroll 2
    for (int k0 = 0; k0 < C - 32; k0 += 32) {
        int kn = k0 + 32;
        // issue next iteration's loads first so they overlap the WMMAs below
        v16bf na0 = load_frag(Ar0 + kn, lhi);
        v16bf na1 = load_frag(Ar1 + kn, lhi);
        v16bf nb0 = load_frag(Br[0] + kn, lhi);
        v16bf nb1 = load_frag(Br[1] + kn, lhi);
        v16bf nb2 = load_frag(Br[2] + kn, lhi);
        v16bf nb3 = load_frag(Br[3] + kn, lhi);
        __builtin_prefetch(Ar0 + kn + 64, 0, 3);   // global_prefetch_b8
        __builtin_prefetch(Br[0] + kn + 64, 0, 3);

        acc[0][0] = wmma_bf16(a0, b0, acc[0][0]);
        acc[0][1] = wmma_bf16(a0, b1, acc[0][1]);
        acc[0][2] = wmma_bf16(a0, b2, acc[0][2]);
        acc[0][3] = wmma_bf16(a0, b3, acc[0][3]);
        acc[1][0] = wmma_bf16(a1, b0, acc[1][0]);
        acc[1][1] = wmma_bf16(a1, b1, acc[1][1]);
        acc[1][2] = wmma_bf16(a1, b2, acc[1][2]);
        acc[1][3] = wmma_bf16(a1, b3, acc[1][3]);

        a0 = na0; a1 = na1;
        b0 = nb0; b1 = nb1; b2 = nb2; b3 = nb3;
    }
    // epilogue K-step
    acc[0][0] = wmma_bf16(a0, b0, acc[0][0]);
    acc[0][1] = wmma_bf16(a0, b1, acc[0][1]);
    acc[0][2] = wmma_bf16(a0, b2, acc[0][2]);
    acc[0][3] = wmma_bf16(a0, b3, acc[0][3]);
    acc[1][0] = wmma_bf16(a1, b0, acc[1][0]);
    acc[1][1] = wmma_bf16(a1, b1, acc[1][1]);
    acc[1][2] = wmma_bf16(a1, b2, acc[1][2]);
    acc[1][3] = wmma_bf16(a1, b3, acc[1][3]);
}

// qkv = x @ W_qkv + b_qkv, split into Q,K (row-major bf16 [T][C]) and
// V stored transposed (bf16 [C][T]) so attention's PV B-fragments are
// contiguous.
__global__ void gemm_qkv(const unsigned short* __restrict__ xb,
                         const unsigned short* __restrict__ wqkvT,
                         const float* __restrict__ bias,
                         unsigned short* __restrict__ qb,
                         unsigned short* __restrict__ kb,
                         unsigned short* __restrict__ vbT) {
    int wave = threadIdx.x >> 5;
    int lane = threadIdx.x & 31;
    int lm = lane & 15, lhi = lane >> 4;
    int tile = blockIdx.x * 4 + wave;
    const int nB = C3 / 64;               // 48 N-strips
    int tm = tile / nB, tn = tile % nB;

    v8f acc[2][4] = {};
    gemm_core_2x4(xb, wqkvT, tm, tn, lm, lhi, acc);

#pragma unroll
    for (int mi = 0; mi < 2; ++mi) {
#pragma unroll
        for (int s = 0; s < 4; ++s) {
            int ncol = tn * 64 + s * 16 + lm;
            float bv = bias[ncol];
#pragma unroll
            for (int v = 0; v < 8; ++v) {
                int r = tm * 32 + mi * 16 + v + 8 * lhi;
                unsigned short hb = f2bf(acc[mi][s][v] + bv);
                if (ncol < C)            qb[(size_t)r * C + ncol] = hb;
                else if (ncol < 2 * C)   kb[(size_t)r * C + (ncol - C)] = hb;
                else                     vbT[(size_t)(ncol - 2 * C) * T + r] = hb;
            }
        }
    }
}

// out = y @ W_proj + b_proj, fp32 output
__global__ void gemm_proj(const unsigned short* __restrict__ yb,
                          const unsigned short* __restrict__ wprojT,
                          const float* __restrict__ bias,
                          float* __restrict__ out) {
    int wave = threadIdx.x >> 5;
    int lane = threadIdx.x & 31;
    int lm = lane & 15, lhi = lane >> 4;
    int tile = blockIdx.x * 4 + wave;
    const int nB = C / 64;                // 16 N-strips
    int tm = tile / nB, tn = tile % nB;

    v8f acc[2][4] = {};
    gemm_core_2x4(yb, wprojT, tm, tn, lm, lhi, acc);

#pragma unroll
    for (int mi = 0; mi < 2; ++mi) {
#pragma unroll
        for (int s = 0; s < 4; ++s) {
            int ncol = tn * 64 + s * 16 + lm;
            float bv = bias[ncol];
#pragma unroll
            for (int v = 0; v < 8; ++v) {
                int r = tm * 32 + mi * 16 + v + 8 * lhi;
                out[(size_t)r * C + ncol] = acc[mi][s][v] + bv;
            }
        }
    }
}

// ---------------------------------------------------------------------------
// Flash attention: one wave per (head, 16-row Q tile). Iterates over 32-wide
// key blocks; K-tile fragments double-buffered across iterations, V-tile
// fragments hoisted ahead of the softmax VALU work, P staged through LDS
// into the A-fragment layout, PV via 4 WMMAs into the running accumulator.
// ---------------------------------------------------------------------------
__global__ void attn_kernel(const unsigned short* __restrict__ qb,
                            const unsigned short* __restrict__ kb,
                            const unsigned short* __restrict__ vbT,
                            unsigned short* __restrict__ yb) {
    __shared__ unsigned short plds[4][16 * 32];   // per-wave 16x32 bf16 P tile

    int wave = threadIdx.x >> 5;
    int lane = threadIdx.x & 31;
    int lm = lane & 15, lhi = lane >> 4;
    int tile = blockIdx.x * 4 + wave;
    const int nQ = T / 16;                // 128 Q tiles per head
    int h  = tile / nQ;
    int tq = tile % nQ;
    const float scale = 0.125f;           // 1/sqrt(64)

    // Q fragments (16 rows x 64 dims = two 16x32 A fragments), loaded once.
    const unsigned short* qrow = qb + (size_t)(tq * 16 + lm) * C + h * DH;
    v16bf qa0 = load_frag(qrow, lhi);
    v16bf qa1 = load_frag(qrow + 32, lhi);

    float mrow[8], lrow[8];
#pragma unroll
    for (int v = 0; v < 8; ++v) { mrow[v] = -INFINITY; lrow[v] = 0.0f; }
    v8f acc[4] = {};                      // 16 rows x 64 dims output accum

    unsigned short* P = plds[wave];
    const unsigned short* kbase = kb + (size_t)lm * C + h * DH;

    // prologue: K fragments for jj = 0
    v16bf kf0 = load_frag(kbase, lhi);
    v16bf kf1 = load_frag(kbase + 32, lhi);
    v16bf kf2 = load_frag(kbase + (size_t)16 * C, lhi);
    v16bf kf3 = load_frag(kbase + (size_t)16 * C + 32, lhi);

    for (int jj = 0; jj < T / 32; ++jj) {
        int j0 = jj * 32;
        int jn = (j0 + 32) & (T - 1);     // wraps to 0 on last iter (discarded)

        // next K tile's fragments: in flight during S WMMAs + softmax
        const unsigned short* kn0 = kbase + (size_t)jn * C;
        v16bf nk0 = load_frag(kn0, lhi);
        v16bf nk1 = load_frag(kn0 + 32, lhi);
        v16bf nk2 = load_frag(kn0 + (size_t)16 * C, lhi);
        v16bf nk3 = load_frag(kn0 + (size_t)16 * C + 32, lhi);

        // ---- S = Q K^T for 32 key rows (two 16x16 tiles) ----
        v8f s0 = {}, s1 = {};
        s0 = wmma_bf16(qa0, kf0, s0);
        s0 = wmma_bf16(qa1, kf1, s0);
        s1 = wmma_bf16(qa0, kf2, s1);
        s1 = wmma_bf16(qa1, kf3, s1);

        // V fragments for this tile: overlap the softmax VALU work below
        v16bf vf[4];
#pragma unroll
        for (int g = 0; g < 4; ++g) {
            const unsigned short* vrow =
                vbT + (size_t)(h * DH + g * 16 + lm) * T + j0;
            vf[g] = load_frag(vrow, lhi);
        }

        // ---- online softmax update; write exp'd P tile to LDS ----
#pragma unroll
        for (int v = 0; v < 8; ++v) {
            float a0 = s0[v] * scale;
            float a1 = s1[v] * scale;
            float mx = fmaxf(a0, a1);
            // reduce across the 16 lanes holding one output row (wave32)
            mx = fmaxf(mx, __shfl_xor(mx, 1, 32));
            mx = fmaxf(mx, __shfl_xor(mx, 2, 32));
            mx = fmaxf(mx, __shfl_xor(mx, 4, 32));
            mx = fmaxf(mx, __shfl_xor(mx, 8, 32));
            float mnew  = fmaxf(mrow[v], mx);
            float alpha = __expf(mrow[v] - mnew);
            float p0 = __expf(a0 - mnew);
            float p1 = __expf(a1 - mnew);
            float ps = p0 + p1;
            ps += __shfl_xor(ps, 1, 32);
            ps += __shfl_xor(ps, 2, 32);
            ps += __shfl_xor(ps, 4, 32);
            ps += __shfl_xor(ps, 8, 32);
            lrow[v] = lrow[v] * alpha + ps;
            mrow[v] = mnew;
#pragma unroll
            for (int g = 0; g < 4; ++g) acc[g][v] = acc[g][v] * alpha;
            int m = v + 8 * lhi;          // C-tile row this value belongs to
            P[m * 32 + lm]      = f2bf(p0);
            P[m * 32 + 16 + lm] = f2bf(p1);
        }
        __syncthreads();                  // P stores visible before fragment loads

        // ---- acc += P(16x32) @ V(32x64) ----
        v16bf pa = load_frag(P + lm * 32, lhi);
        acc[0] = wmma_bf16(pa, vf[0], acc[0]);
        acc[1] = wmma_bf16(pa, vf[1], acc[1]);
        acc[2] = wmma_bf16(pa, vf[2], acc[2]);
        acc[3] = wmma_bf16(pa, vf[3], acc[3]);
        // WAR on P vs next iteration's stores is safe: per-wave LDS slice and
        // the LDS pipe keeps same-wave DS ops in order.

        kf0 = nk0; kf1 = nk1; kf2 = nk2; kf3 = nk3;
    }

    // ---- normalize and write y (bf16, row-major [T][C]) ----
#pragma unroll
    for (int g = 0; g < 4; ++g) {
#pragma unroll
        for (int v = 0; v < 8; ++v) {
            int m = v + 8 * lhi;
            float o = acc[g][v] / lrow[v];
            yb[(size_t)(tq * 16 + m) * C + h * DH + g * 16 + lm] = f2bf(o);
        }
    }
}

// ---------------------------------------------------------------------------
// Host-side launcher
// ---------------------------------------------------------------------------
extern "C" void kernel_launch(void* const* d_in, const int* in_sizes, int n_in,
                              void* d_out, int out_size, void* d_ws, size_t ws_size,
                              hipStream_t stream) {
    const float* x      = (const float*)d_in[0];   // [T][C]
    const float* W_qkv  = (const float*)d_in[1];   // [C][3C]
    const float* b_qkv  = (const float*)d_in[2];   // [3C]
    const float* W_proj = (const float*)d_in[3];   // [C][C]
    const float* b_proj = (const float*)d_in[4];   // [C]
    float* out = (float*)d_out;                    // [T][C]

    // workspace layout (bf16 = unsigned short)
    unsigned short* w = (unsigned short*)d_ws;
    unsigned short* xb     = w;            w += (size_t)T * C;    // x bf16
    unsigned short* wqkvT  = w;            w += (size_t)C3 * C;   // W_qkv^T bf16
    unsigned short* wprojT = w;            w += (size_t)C * C;    // W_proj^T bf16
    unsigned short* qb     = w;            w += (size_t)T * C;    // Q bf16 [T][C]
    unsigned short* kb     = w;            w += (size_t)T * C;    // K bf16 [T][C]
    unsigned short* vbT    = w;            w += (size_t)C * T;    // V^T bf16 [C][T]
    unsigned short* yb     = w;            w += (size_t)T * C;    // attn out bf16

    const int thr = 256;
    cvt_f32_bf16<<<(T * C + thr - 1) / thr, thr, 0, stream>>>(x, xb, T * C);
    cvt_transpose_f32_bf16<<<(C * C3 + thr - 1) / thr, thr, 0, stream>>>(
        W_qkv, wqkvT, C, C3);
    cvt_transpose_f32_bf16<<<(C * C + thr - 1) / thr, thr, 0, stream>>>(
        W_proj, wprojT, C, C);

    // (T/32)*(C3/64) = 3072 wave-tiles, 4 waves (128 threads) per block
    gemm_qkv<<<(T / 32) * (C3 / 64) / 4, 128, 0, stream>>>(
        xb, wqkvT, b_qkv, qb, kb, vbT);

    // H*(T/16) = 2048 wave-tiles
    attn_kernel<<<H * (T / 16) / 4, 128, 0, stream>>>(qb, kb, vbT, yb);

    // (T/32)*(C/64) = 1024 wave-tiles
    gemm_proj<<<(T / 32) * (C / 64) / 4, 128, 0, stream>>>(
        yb, wprojT, b_proj, out);
}
